// SAGEEncoder_6786048328256
// MI455X (gfx1250) — matrix operand order repaired and checked by reading, not compiled, not run on hardware
//
#include <hip/hip_runtime.h>

#define N_NODES 100000
#define N_EDGES 1600000
#define IN_DIM  32
#define HID_DIM 64
#define EMB_DIM 128
#define N_GRAPHS 512
#define M_TILES (N_NODES / 16)   // 6250, exact

typedef __attribute__((ext_vector_type(16))) __bf16 v16bf;
typedef __attribute__((ext_vector_type(8)))  __bf16 v8bf;
typedef __attribute__((ext_vector_type(8)))  float  v8f;

// fp32 -> bf16 round-to-nearest-even (bitwise)
__device__ __forceinline__ __bf16 f2bf(float f) {
    unsigned u = __builtin_bit_cast(unsigned, f);
    u += 0x7FFFu + ((u >> 16) & 1u);
    unsigned short h = (unsigned short)(u >> 16);
    return __builtin_bit_cast(__bf16, h);
}
// bf16 -> fp32 (exact widen)
__device__ __forceinline__ float bf2f(__bf16 b) {
    unsigned short h = __builtin_bit_cast(unsigned short, b);
    return __builtin_bit_cast(float, (unsigned)h << 16);
}

// ---------------- zero workspace accumulators ----------------
__global__ void zero_kernel(float* __restrict__ p, long n) {
    long i = (long)blockIdx.x * blockDim.x + threadIdx.x;
    long stride = (long)gridDim.x * blockDim.x;
    for (; i < n; i += stride) p[i] = 0.0f;
}

// ---------------- fp32 -> bf16 streaming convert ----------------
__global__ void cvt_kernel(const float* __restrict__ src, __bf16* __restrict__ dst, long n) {
    long i = (long)blockIdx.x * blockDim.x + threadIdx.x;
    if (i < n) dst[i] = f2bf(src[i]);
}

// ---------------- weight transpose+convert: W[k][n] -> Wt[n][k] bf16 ----------------
__global__ void wtrans_kernel(const float* __restrict__ W, __bf16* __restrict__ Wt,
                              int K, int N) {
    int i = (int)(blockIdx.x * blockDim.x + threadIdx.x);
    if (i >= K * N) return;
    int k = i / N, n = i - k * N;
    Wt[n * K + k] = f2bf(W[i]);
}

// ---------------- normalize (mean) + convert: out = bf16(agg * 1/max(deg,1)) ----------
__global__ void normcvt_kernel(const float* __restrict__ agg, const float* __restrict__ deg,
                               __bf16* __restrict__ out, int dim_shift, long n) {
    long i = (long)blockIdx.x * blockDim.x + threadIdx.x;
    if (i >= n) return;
    long node = i >> dim_shift;
    out[i] = f2bf(agg[i] * (1.0f / fmaxf(deg[node], 1.0f)));
}

// ---------------- edge scatter, layer 1 (32 feats: 1 lane each) ----------------
__global__ void scatter1_kernel(const float* __restrict__ x,
                                const long long* __restrict__ ei,
                                float* __restrict__ agg1,
                                float* __restrict__ deg) {
    int e = (int)((blockIdx.x * blockDim.x + threadIdx.x) >> 5);
    int lane = (int)(threadIdx.x & 31);
    if (e >= N_EDGES) return;
    long long s = ei[e];
    long long d = ei[N_EDGES + e];
    atomicAdd(&agg1[d * IN_DIM + lane], x[s * IN_DIM + lane]);
    if (lane == 0) atomicAdd(&deg[d], 1.0f);
}

// ---------------- edge scatter, layer 2 (64 feats from bf16 h1) ----------------
__global__ void scatter2_kernel(const __bf16* __restrict__ h1b,
                                const long long* __restrict__ ei,
                                float* __restrict__ agg2) {
    int e = (int)((blockIdx.x * blockDim.x + threadIdx.x) >> 5);
    int lane = (int)(threadIdx.x & 31);
    if (e >= N_EDGES) return;
    long long s = ei[e];
    long long d = ei[N_EDGES + e];
    atomicAdd(&agg2[d * HID_DIM + lane],      bf2f(h1b[s * HID_DIM + lane]));
    atomicAdd(&agg2[d * HID_DIM + lane + 32], bf2f(h1b[s * HID_DIM + lane + 32]));
}

// ---------------- per-graph node counts ----------------
__global__ void count_kernel(const long long* __restrict__ batch,
                             float* __restrict__ pcnt) {
    int i = (int)(blockIdx.x * blockDim.x + threadIdx.x);
    if (i >= N_NODES) return;
    atomicAdd(&pcnt[(int)batch[i]], 1.0f);
}

// Build a 16-bit A fragment (16x32 tile): per lane one contiguous
// 16-element K-run at kbase+2*lo (K = 16h..16h+15) -> two b128 loads + shuffle.
__device__ __forceinline__ v16bf load_a_frag(const __bf16* __restrict__ row, int kbase, int lo) {
    v8bf a = *(const v8bf*)(row + kbase + 2 * lo);
    v8bf b = *(const v8bf*)(row + kbase + 2 * lo + 8);
    return __builtin_shufflevector(a, b, 0, 1, 2, 3, 4, 5, 6, 7, 8, 9, 10, 11, 12, 13, 14, 15);
}

// ---------------- layer 1: h1b = bf16(relu((agg1/deg)W1l + x W1r + b1)) ----------------
__global__ void gemm1_kernel(const __bf16* __restrict__ a1b,
                             const __bf16* __restrict__ xb,
                             const __bf16* __restrict__ Wlt,   // [HID_DIM][IN_DIM]
                             const __bf16* __restrict__ Wrt,
                             const float* __restrict__ bias,
                             __bf16* __restrict__ h1b) {
    int wave = (int)((blockIdx.x * blockDim.x + threadIdx.x) >> 5);
    if (wave >= M_TILES) return;                 // wave-uniform guard: EXEC all-1s inside
    int lane = (int)(threadIdx.x & 31);
    int m0 = wave * 16;
    int m  = m0 + (lane & 15);
    int lo = (lane & 16) ? 8 : 0;

    v16bf a_agg = load_a_frag(a1b + (size_t)m * IN_DIM, 0, lo);
    v16bf a_x   = load_a_frag(xb  + (size_t)m * IN_DIM, 0, lo);

    int klo   = (lane & 16) ? 16 : 0;            // B 32x16: lanes 0-15 K=0-15, 16-31 K=16-31
    int nl    = lane & 15;
    int rbase = m0 + ((lane & 16) ? 8 : 0);      // C/D: vgpr g -> M = g (+8 in hi half)

#pragma unroll
    for (int jn = 0; jn < HID_DIM / 16; ++jn) {
        int n = jn * 16 + nl;
        v16bf bl = *(const v16bf*)(Wlt + (size_t)n * IN_DIM + klo);
        v16bf br = *(const v16bf*)(Wrt + (size_t)n * IN_DIM + klo);
        v8f c = {};
        c = __builtin_amdgcn_wmma_f32_16x16x32_bf16(false, a_agg, false, bl, (short)0, c, false, false);
        c = __builtin_amdgcn_wmma_f32_16x16x32_bf16(false, a_x,   false, br, (short)0, c, false, false);
        float bv = bias[n];
#pragma unroll
        for (int g = 0; g < 8; ++g) {
            float v = fmaxf(c[g] + bv, 0.0f);
            h1b[(size_t)(rbase + g) * HID_DIM + n] = f2bf(v);
        }
    }
}

// ---------------- layer 2 + fused mean-pool scatter ----------------
__global__ void gemm2_pool_kernel(const __bf16* __restrict__ a2b,
                                  const __bf16* __restrict__ h1b,
                                  const __bf16* __restrict__ Wlt,  // [EMB_DIM][HID_DIM]
                                  const __bf16* __restrict__ Wrt,
                                  const float* __restrict__ bias,
                                  const long long* __restrict__ batch,
                                  float* __restrict__ psum) {
    int wave = (int)((blockIdx.x * blockDim.x + threadIdx.x) >> 5);
    if (wave >= M_TILES) return;
    int lane = (int)(threadIdx.x & 31);
    int m0 = wave * 16;
    int m  = m0 + (lane & 15);
    int lo = (lane & 16) ? 8 : 0;

    v16bf a_agg[2], a_h[2];
#pragma unroll
    for (int kk = 0; kk < 2; ++kk) {
        a_agg[kk] = load_a_frag(a2b + (size_t)m * HID_DIM, kk * 32, lo);
        a_h[kk]   = load_a_frag(h1b + (size_t)m * HID_DIM, kk * 32, lo);
    }
    int klo   = (lane & 16) ? 16 : 0;
    int nl    = lane & 15;
    int rbase = m0 + ((lane & 16) ? 8 : 0);
    int barr[8];
#pragma unroll
    for (int g = 0; g < 8; ++g) barr[g] = (int)batch[rbase + g];

#pragma unroll
    for (int jn = 0; jn < EMB_DIM / 16; ++jn) {
        int n = jn * 16 + nl;
        v8f c = {};
#pragma unroll
        for (int kk = 0; kk < 2; ++kk) {
            v16bf bl = *(const v16bf*)(Wlt + (size_t)n * HID_DIM + kk * 32 + klo);
            v16bf br = *(const v16bf*)(Wrt + (size_t)n * HID_DIM + kk * 32 + klo);
            c = __builtin_amdgcn_wmma_f32_16x16x32_bf16(false, a_agg[kk], false, bl, (short)0, c, false, false);
            c = __builtin_amdgcn_wmma_f32_16x16x32_bf16(false, a_h[kk],   false, br, (short)0, c, false, false);
        }
        float bv = bias[n];
#pragma unroll
        for (int g = 0; g < 8; ++g) {
            float v = fmaxf(c[g] + bv, 0.0f);
            atomicAdd(&psum[barr[g] * EMB_DIM + n], v);
        }
    }
}

// ---------------- finalize: out = psum / max(pcnt,1) ----------------
__global__ void finalize_kernel(const float* __restrict__ psum,
                                const float* __restrict__ pcnt,
                                float* __restrict__ out) {
    int i = (int)(blockIdx.x * blockDim.x + threadIdx.x);
    if (i >= N_GRAPHS * EMB_DIM) return;
    out[i] = psum[i] / fmaxf(pcnt[i >> 7], 1.0f);
}

extern "C" void kernel_launch(void* const* d_in, const int* in_sizes, int n_in,
                              void* d_out, int out_size, void* d_ws, size_t ws_size,
                              hipStream_t stream) {
    const float*     x     = (const float*)d_in[0];
    const long long* ei    = (const long long*)d_in[1];   // [2, E] int64
    // d_in[2] = edge_weight (unused by reference)
    const long long* batch = (const long long*)d_in[3];   // int64
    const float* W1l = (const float*)d_in[4];
    const float* W1r = (const float*)d_in[5];
    const float* b1  = (const float*)d_in[6];
    const float* W2l = (const float*)d_in[7];
    const float* W2r = (const float*)d_in[8];
    const float* b2  = (const float*)d_in[9];
    float* out = (float*)d_out;

    // ---- fp32 region (atomic accumulators; zeroed every launch) ----
    float* ws   = (float*)d_ws;
    float* agg1 = ws;                                     // N*32
    float* deg  = agg1 + (size_t)N_NODES * IN_DIM;        // N
    float* agg2 = deg  + (size_t)N_NODES;                 // N*64
    float* psum = agg2 + (size_t)N_NODES * HID_DIM;       // 512*128
    float* pcnt = psum + (size_t)N_GRAPHS * EMB_DIM;      // 512
    long zeroN = (long)N_NODES * IN_DIM + N_NODES + (long)N_NODES * HID_DIM
               + (long)N_GRAPHS * EMB_DIM + N_GRAPHS;     // = 9,766,048 floats (32B-multiple)

    // ---- bf16 region (operands pre-laid-out for WMMA fragments) ----
    __bf16* bbase = (__bf16*)(pcnt + N_GRAPHS);
    __bf16* xb   = bbase;                                 // N*32
    __bf16* a1b  = xb   + (size_t)N_NODES * IN_DIM;       // N*32
    __bf16* h1b  = a1b  + (size_t)N_NODES * IN_DIM;       // N*64
    __bf16* a2b  = h1b  + (size_t)N_NODES * HID_DIM;      // N*64
    __bf16* w1lt = a2b  + (size_t)N_NODES * HID_DIM;      // 64*32
    __bf16* w1rt = w1lt + HID_DIM * IN_DIM;               // 64*32
    __bf16* w2lt = w1rt + HID_DIM * IN_DIM;               // 128*64
    __bf16* w2rt = w2lt + EMB_DIM * HID_DIM;              // 128*64

    zero_kernel<<<2048, 256, 0, stream>>>(ws, zeroN);

    // prep: convert x, transpose+convert weights (tiny, L2-resident)
    cvt_kernel<<<(N_NODES * IN_DIM + 255) / 256, 256, 0, stream>>>(x, xb, (long)N_NODES * IN_DIM);
    wtrans_kernel<<<(IN_DIM * HID_DIM + 255) / 256, 256, 0, stream>>>(W1l, w1lt, IN_DIM, HID_DIM);
    wtrans_kernel<<<(IN_DIM * HID_DIM + 255) / 256, 256, 0, stream>>>(W1r, w1rt, IN_DIM, HID_DIM);
    wtrans_kernel<<<(HID_DIM * EMB_DIM + 255) / 256, 256, 0, stream>>>(W2l, w2lt, HID_DIM, EMB_DIM);
    wtrans_kernel<<<(HID_DIM * EMB_DIM + 255) / 256, 256, 0, stream>>>(W2r, w2rt, HID_DIM, EMB_DIM);

    scatter1_kernel<<<N_EDGES / 8, 256, 0, stream>>>(x, ei, agg1, deg);

    normcvt_kernel<<<(N_NODES * IN_DIM + 255) / 256, 256, 0, stream>>>(
        agg1, deg, a1b, 5, (long)N_NODES * IN_DIM);

    gemm1_kernel<<<(M_TILES + 7) / 8, 256, 0, stream>>>(a1b, xb, w1lt, w1rt, b1, h1b);

    scatter2_kernel<<<N_EDGES / 8, 256, 0, stream>>>(h1b, ei, agg2);

    normcvt_kernel<<<(N_NODES * HID_DIM + 255) / 256, 256, 0, stream>>>(
        agg2, deg, a2b, 6, (long)N_NODES * HID_DIM);

    count_kernel<<<(N_NODES + 255) / 256, 256, 0, stream>>>(batch, pcnt);

    gemm2_pool_kernel<<<(M_TILES + 7) / 8, 256, 0, stream>>>(
        a2b, h1b, w2lt, w2rt, b2, batch, psum);

    finalize_kernel<<<(N_GRAPHS * EMB_DIM + 255) / 256, 256, 0, stream>>>(psum, pcnt, out);
}
